// FloMo_49821620634127
// MI455X (gfx1250) — matrix operand order repaired and checked by reading, not compiled, not run on hardware
//
#include <hip/hip_runtime.h>
#include <hip/hip_bf16.h>

// CDNA5 / gfx1250: wave32, WMMA 16x16x32 bf16 -> f32 accum.

typedef __attribute__((ext_vector_type(16))) __bf16       v16bf;
typedef __attribute__((ext_vector_type(8)))  float        v8f;
typedef __attribute__((ext_vector_type(4)))  unsigned int u32x4;  // builtin vec: AS-safe

#define LDSAS __attribute__((address_space(3)))
#define GLBAS __attribute__((address_space(1)))

#define NROWS   32     // rows per workgroup (2 row-groups x 16)
#define HID     512
#define NIN     12
#define NC      16
#define DHALF   6
#define DIN     22     // DHALF + NC
#define DOUT    138    // (3*8-1)*6
#define NLAYER  8
#define BVAL    3.0f
#define MINW    0.001f
#define MINH    0.001f
#define MIND    0.001f

union Frag16 { v16bf v; u32x4 u[2]; };

// Packed-B fragment from global: 16 contiguous bf16 per lane (32B) -> 2x global_load_b128.
__device__ inline v16bf load_frag_b(const GLBAS __bf16* p) {
  Frag16 f;
  f.u[0] = *(const GLBAS u32x4*)(p);
  f.u[1] = *(const GLBAS u32x4*)(p + 8);
  return f.v;
}

// A fragment from LDS: lane holds row (lane&15); K chunks at klo and klo+16.
// Matches ISA 16-bit A layout: lanes0-15 K={0..7,16..23}, lanes16-31 K={8..15,24..31}.
__device__ inline v16bf load_frag_a(const LDSAS __bf16* p) {
  Frag16 f;
  f.u[0] = *(const LDSAS u32x4*)(p);
  f.u[1] = *(const LDSAS u32x4*)(p + 16);
  return f.v;
}

// Branch-free ELU: native v_exp_f32 + cndmask (no exec-mask divergence).
__device__ inline float elu_fast(float x) {
  float e = __expf(x) - 1.f;
  return x > 0.f ? x : e;
}

// 16xK @ Kx512 GEMM + bias + ELU -> bf16 LDS. One wave: 16 rows x 128 cols (8 tiles).
template <int KT>
__device__ inline void dense512(const LDSAS __bf16* A, int apitch,
                                const GLBAS __bf16* Bp,
                                const GLBAS float*  bias,
                                LDSAS __bf16* H, int lane, int rg, int cg) {
  const int lrow = lane & 15;
  const int lhi  = lane >> 4;
  v8f acc[8] = {};
  const LDSAS __bf16* arow = A + (rg * 16 + lrow) * apitch + lhi * 8;
#pragma unroll
  for (int kt = 0; kt < KT; ++kt) {
    v16bf af = load_frag_a(arow + kt * 32);
    const GLBAS __bf16* bb = Bp + (((size_t)kt * 32 + cg * 8) * 32 + lane) * 16;
#pragma unroll
    for (int t = 0; t < 8; ++t) {
      v16bf bfv = load_frag_b(bb + (size_t)t * 512);   // 32 lanes * 16 elems per tile
      acc[t] = __builtin_amdgcn_wmma_f32_16x16x32_bf16(
          false, af, false, bfv, (short)0, acc[t], false, false);
    }
  }
#pragma unroll
  for (int t = 0; t < 8; ++t) {
    int n = (cg * 8 + t) * 16 + lrow;
    float bv = bias[n];
#pragma unroll
    for (int v = 0; v < 8; ++v) {
      int r = rg * 16 + lhi * 8 + v;   // C layout: VGPR v -> M = v (+8 for hi lanes)
      float x = elu_fast(acc[t][v] + bv);
      H[r * HID + n] = (__bf16)x;
    }
  }
}

// 16x512 @ 512x138(padded 144) GEMM + bias -> f32 LDS. 9 n-tiles over 4 col-groups.
__device__ inline void head_layer(const LDSAS __bf16* A,
                                  const GLBAS __bf16* Bp,
                                  const GLBAS float*  bias,
                                  LDSAS float* ob, int lane, int rg, int cg) {
  const int lrow = lane & 15, lhi = lane >> 4;
  v8f acc[3] = {};
  int ntile[3]; int cnt = 0;
  for (int nt = cg; nt < 9; nt += 4) ntile[cnt++] = nt;
  const LDSAS __bf16* arow = A + (rg * 16 + lrow) * HID + lhi * 8;
  for (int kt = 0; kt < 16; ++kt) {
    v16bf af = load_frag_a(arow + kt * 32);
    for (int ti = 0; ti < cnt; ++ti) {
      const GLBAS __bf16* bb = Bp + (((size_t)kt * 9 + ntile[ti]) * 32 + lane) * 16;
      v16bf bfv = load_frag_b(bb);
      acc[ti] = __builtin_amdgcn_wmma_f32_16x16x32_bf16(
          false, af, false, bfv, (short)0, acc[ti], false, false);
    }
  }
  for (int ti = 0; ti < cnt; ++ti) {
    int n = ntile[ti] * 16 + lrow;
    if (n < DOUT) {
      float bv = bias[n];
#pragma unroll
      for (int v = 0; v < 8; ++v) {
        int r = rg * 16 + lhi * 8 + v;
        ob[r * 144 + n] = acc[ti][v] + bv;
      }
    }
  }
}

// Pack f32 row-major [L][Ks][Ns] weights into bf16 WMMA fragment tiles:
// element ((l*KT+kt)*NT+nt)*32*16 + lane*16 + j  holds
//   W[l][kt*32 + (lane>=16?8:0) + (j<8?j:j+8)][nt*16 + lane%16]   (zero-padded OOB)
__global__ void pack_kernel(const float* __restrict__ W, __bf16* __restrict__ Wp,
                            int Lw, int Ks, int Ns, int KT, int NT) {
  const GLBAS float*  Wg  = (const GLBAS float*)W;
  GLBAS __bf16*       Wpg = (GLBAS __bf16*)Wp;
  long total = (long)Lw * KT * NT * 32;
  for (long idx = (long)blockIdx.x * blockDim.x + threadIdx.x; idx < total;
       idx += (long)gridDim.x * blockDim.x) {
    int  lane = (int)(idx & 31);
    long rest = idx >> 5;
    int  nt = (int)(rest % NT); rest /= NT;
    int  kt = (int)(rest % KT);
    int  l  = (int)(rest / KT);
    int n = nt * 16 + (lane & 15);
    int kbase = kt * 32 + (lane >> 4) * 8;
    GLBAS __bf16* dst = Wpg + idx * 16;
    const GLBAS float* src = Wg + (size_t)l * Ks * Ns;
#pragma unroll
    for (int j = 0; j < 16; ++j) {
      int k = kbase + (j < 8 ? j : j + 8);
      float v = (k < Ks && n < Ns) ? src[(size_t)k * Ns + n] : 0.f;
      dst[j] = (__bf16)v;
    }
  }
}

__global__ __launch_bounds__(256) void flomo_kernel(
    const float* __restrict__ z,  const float* __restrict__ c,
    const float* __restrict__ b0, const float* __restrict__ b1,
    const float* __restrict__ b2, const float* __restrict__ b3,
    const int*   __restrict__ perms,
    const __bf16* __restrict__ W0p, const __bf16* __restrict__ W1p,
    const __bf16* __restrict__ W2p, const __bf16* __restrict__ W3p,
    float* __restrict__ out, int Ntot) {
  __shared__ __align__(16) float  xs[NROWS][NIN];
  __shared__ __align__(16) float  xt[NROWS][NIN];
  __shared__ __align__(16) float  cs[NROWS][NC];
  __shared__ __align__(16) float  ldt[NROWS];
  __shared__ __align__(16) __bf16 inp[NROWS][32];
  __shared__ __align__(16) __bf16 hA[NROWS][HID];
  __shared__ __align__(16) __bf16 hB[NROWS][HID];

  // Explicit LDS-address-space views (fold to no-op casts; no null-check cndmasks).
  LDSAS __bf16* inpL = (LDSAS __bf16*)&inp[0][0];
  LDSAS __bf16* hAL  = (LDSAS __bf16*)&hA[0][0];
  LDSAS __bf16* hBL  = (LDSAS __bf16*)&hB[0][0];
  LDSAS float*  obL  = (LDSAS float*)&hB[0][0];  // overlay: 32*144*4 <= 32KB, hB free then
  LDSAS float*  xsL  = (LDSAS float*)&xs[0][0];

  // Explicit global-address-space views.
  const GLBAS __bf16* W0g = (const GLBAS __bf16*)W0p;
  const GLBAS __bf16* W1g = (const GLBAS __bf16*)W1p;
  const GLBAS __bf16* W2g = (const GLBAS __bf16*)W2p;
  const GLBAS __bf16* W3g = (const GLBAS __bf16*)W3p;
  const GLBAS float*  b0g = (const GLBAS float*)b0;
  const GLBAS float*  b1g = (const GLBAS float*)b1;
  const GLBAS float*  b2g = (const GLBAS float*)b2;
  const GLBAS float*  b3g = (const GLBAS float*)b3;

  const int tid  = threadIdx.x;
  const int row0 = blockIdx.x * NROWS;

  for (int i = tid; i < NROWS * NIN; i += 256)
    xs[i / NIN][i % NIN] = z[(size_t)(row0 + i / NIN) * NIN + i % NIN];
  for (int i = tid; i < NROWS * NC; i += 256)
    cs[i / NC][i % NC] = c[(size_t)(row0 + i / NC) * NC + i % NC];
  if (tid < NROWS) ldt[tid] = 0.f;
  __syncthreads();

  const int lane = tid & 31;
  const int wv   = tid >> 5;
  const int rg   = wv >> 2;   // row-group 0..1
  const int cg   = wv & 3;    // col-group 0..3

  for (int li = 0; li < NLAYER; ++li) {
    // build inp = concat(x1, c), zero-padded to K=32, as bf16
    for (int i = tid; i < NROWS * 32; i += 256) {
      int r = i >> 5, k = i & 31;
      float v = (k < DHALF) ? xs[r][k] : ((k < DIN) ? cs[r][k - DHALF] : 0.f);
      inpL[i] = (__bf16)v;
    }
    __syncthreads();
    dense512<1>(inpL, 32, W0g + (size_t)li * 32 * 512, b0g + li * HID,
                hAL, lane, rg, cg);
    __syncthreads();
    dense512<16>(hAL, HID, W1g + (size_t)li * 16 * 32 * 512, b1g + li * HID,
                 hBL, lane, rg, cg);
    __syncthreads();
    dense512<16>(hBL, HID, W2g + (size_t)li * 16 * 32 * 512, b2g + li * HID,
                 hAL, lane, rg, cg);
    __syncthreads();
    head_layer(hAL, W3g + (size_t)li * 16 * 9 * 512, b3g + li * DOUT,
               obL, lane, rg, cg);
    __syncthreads();

    // rational-quadratic spline on x2 (6 dims), scalar per (row, dim)
    {
      int r = tid >> 3, dm = tid & 7;
      if (dm < (NIN - DHALF)) {
        const LDSAS float* o = obL + r * 144 + dm * 23;
        float cw[9], chh[9], dd[9];
        {
          float mx = o[0];
          for (int k = 1; k < 8; ++k) mx = fmaxf(mx, o[k]);
          float s = 0.f, e[8];
          for (int k = 0; k < 8; ++k) { e[k] = __expf(o[k] - mx); s += e[k]; }
          float inv = 1.f / s, run = 0.f;
          cw[0] = -BVAL;
          for (int k = 0; k < 8; ++k) {
            run += MINW + (1.f - MINW * 8.f) * (e[k] * inv);
            cw[k + 1] = 2.f * BVAL * run - BVAL;
          }
          cw[8] = BVAL;
        }
        {
          float mx = o[8];
          for (int k = 1; k < 8; ++k) mx = fmaxf(mx, o[8 + k]);
          float s = 0.f, e[8];
          for (int k = 0; k < 8; ++k) { e[k] = __expf(o[8 + k] - mx); s += e[k]; }
          float inv = 1.f / s, run = 0.f;
          chh[0] = -BVAL;
          for (int k = 0; k < 8; ++k) {
            run += MINH + (1.f - MINH * 8.f) * (e[k] * inv);
            chh[k + 1] = 2.f * BVAL * run - BVAL;
          }
          chh[8] = BVAL;
        }
        // d = MIN_D + softplus(ud); edges: MIN_D + softplus(DERIV_CONST) == 1 exactly
        dd[0] = 1.f; dd[8] = 1.f;
        for (int k = 1; k < 8; ++k) {
          float u = o[16 + (k - 1)];
          // stable branch-free softplus: max(u,0) + log1p(exp(-|u|))
          float sp = fmaxf(u, 0.f) + __logf(1.f + __expf(-fabsf(u)));
          dd[k] = MIND + sp;
        }
        float x2 = xsL[r * NIN + DHALF + dm];
        float xc = fminf(fmaxf(x2, -BVAL), BVAL);
        int idx = 0;
        for (int k = 1; k < 8; ++k) idx += (xc >= cw[k]) ? 1 : 0;
        idx = idx > 7 ? 7 : idx;
        float icw = cw[idx], iw = cw[idx + 1] - cw[idx];
        float ich = chh[idx], ih = chh[idx + 1] - chh[idx];
        float idelta = ih / iw;
        float id0 = dd[idx], id1 = dd[idx + 1];
        float th = (xc - icw) / iw;
        float t1m = th * (1.f - th);
        float num = ih * (idelta * th * th + id0 * t1m);
        float den = idelta + (id0 + id1 - 2.f * idelta) * t1m;
        float ov  = ich + num / den;
        float dnum = idelta * idelta *
                     (id1 * th * th + 2.f * idelta * t1m + id0 * (1.f - th) * (1.f - th));
        float lad = __logf(dnum) - 2.f * __logf(den);
        bool inside = (x2 >= -BVAL) && (x2 <= BVAL);
        xsL[r * NIN + DHALF + dm] = inside ? ov : x2;
        if (inside) atomicAdd(&ldt[r], lad);
      }
    }
    __syncthreads();

    if (li < NLAYER - 1) {
      for (int i = tid; i < NROWS * NIN; i += 256)
        xt[i / NIN][i % NIN] = xs[i / NIN][i % NIN];
      __syncthreads();
      for (int i = tid; i < NROWS * NIN; i += 256) {
        int r = i / NIN, j = i % NIN;
        xs[r][j] = xt[r][perms[li * NIN + j]];
      }
      __syncthreads();
    }
  }

  for (int i = tid; i < NROWS * NIN; i += 256)
    out[(size_t)(row0 + i / NIN) * NIN + i % NIN] = xs[i / NIN][i % NIN];
  if (tid < NROWS) out[(size_t)Ntot * NIN + row0 + tid] = ldt[tid];
}

extern "C" void kernel_launch(void* const* d_in, const int* in_sizes, int n_in,
                              void* d_out, int out_size, void* d_ws, size_t ws_size,
                              hipStream_t stream) {
  const float* z  = (const float*)d_in[0];
  const float* c  = (const float*)d_in[1];
  const float* W0 = (const float*)d_in[2];
  const float* b0 = (const float*)d_in[3];
  const float* W1 = (const float*)d_in[4];
  const float* b1 = (const float*)d_in[5];
  const float* W2 = (const float*)d_in[6];
  const float* b2 = (const float*)d_in[7];
  const float* W3 = (const float*)d_in[8];
  const float* b3 = (const float*)d_in[9];
  const int* perms = (const int*)d_in[10];

  __bf16* ws = (__bf16*)d_ws;
  const size_t szW0 = (size_t)8 * 1 * 32 * 512;    // bf16 elements
  const size_t szW1 = (size_t)8 * 16 * 32 * 512;
  __bf16* W0p = ws;
  __bf16* W1p = W0p + szW0;
  __bf16* W2p = W1p + szW1;
  __bf16* W3p = W2p + szW1;

  // Pack weights to bf16 WMMA fragment layout (runs every call; deterministic).
  long t0 = 8L * 1 * 32 * 32;
  pack_kernel<<<(int)((t0 + 255) / 256), 256, 0, stream>>>(W0, W0p, 8, 22, 512, 1, 32);
  long t1 = 8L * 16 * 32 * 32;
  pack_kernel<<<(int)((t1 + 255) / 256), 256, 0, stream>>>(W1, W1p, 8, 512, 512, 16, 32);
  pack_kernel<<<(int)((t1 + 255) / 256), 256, 0, stream>>>(W2, W2p, 8, 512, 512, 16, 32);
  long t3 = 8L * 16 * 9 * 32;
  pack_kernel<<<(int)((t3 + 255) / 256), 256, 0, stream>>>(W3, W3p, 8, 512, 138, 16, 9);

  int Ntot = in_sizes[0] / NIN;
  flomo_kernel<<<Ntot / NROWS, 256, 0, stream>>>(z, c, b0, b1, b2, b3, perms,
                                                 W0p, W1p, W2p, W3p,
                                                 (float*)d_out, Ntot);
}
